// GIN_31447750541888
// MI455X (gfx1250) — matrix operand order repaired and checked by reading, not compiled, not run on hardware
//
#include <hip/hip_runtime.h>
#include <hip/hip_bf16.h>
#include <cstdint>

// GIN (3-layer) on MI455X / gfx1250.
// f16 WMMA (f32 accumulate) for all GEMMs, weights staged to LDS via
// GLOBAL_LOAD_ASYNC_TO_LDS_B128 (ASYNCcnt), hidden activations pass through
// LDS as f16, pooling fused via f32 atomics, sorted-run segment sums.

#define NN   100000
#define NE   1600000
#define FEAT 128
#define HID  128
#define TGT  64
#define NL   3
#define NG   512

typedef __attribute__((ext_vector_type(16))) _Float16 v16h;
typedef __attribute__((ext_vector_type(8)))  _Float16 v8h;
typedef __attribute__((ext_vector_type(8)))  float    v8f;

static __device__ __forceinline__ v16h cat16(v8h lo, v8h hi) {
  return __builtin_shufflevector(lo, hi, 0,1,2,3,4,5,6,7,8,9,10,11,12,13,14,15);
}
static __device__ __forceinline__ v8h cvt8(float4 a, float4 b) {
  v8h r;
  r[0]=(_Float16)a.x; r[1]=(_Float16)a.y; r[2]=(_Float16)a.z; r[3]=(_Float16)a.w;
  r[4]=(_Float16)b.x; r[5]=(_Float16)b.y; r[6]=(_Float16)b.z; r[7]=(_Float16)b.w;
  return r;
}
// A fragment (16x32 f16) from a global f32 row. Layout: lanes 0-15 hold
// K [kb..kb+15]; lanes 16-31 hold [kb+16..kb+31].
static __device__ __forceinline__ v16h load_a_f32(const float* __restrict__ row,
                                                  int kbase, int hi) {
  const float4* p0 = (const float4*)(row + kbase + hi * 16);
  return cat16(cvt8(p0[0], p0[1]), cvt8(p0[2], p0[3]));
}
// Same A fragment from an LDS f16 row.
static __device__ __forceinline__ v16h load_a_lds(const _Float16* row,
                                                  int kbase, int hi) {
  v8h lo = *(const v8h*)(row + kbase + hi * 16);
  v8h up = *(const v8h*)(row + kbase + hi * 16 + 8);
  return cat16(lo, up);
}

// Async global->LDS copy, 16B per lane per issue (dynamic-LDS base is 0).
static __device__ __forceinline__ void async_copy_16(unsigned lds_base,
                                                     const char* gsrc, int bytes,
                                                     int tid) {
  for (int off = tid * 16; off < bytes; off += 256 * 16) {
    unsigned la = lds_base + (unsigned)off;
    unsigned long long ga = (unsigned long long)(uintptr_t)(gsrc + off);
    asm volatile("global_load_async_to_lds_b128 %0, %1, off"
                 :: "v"(la), "v"(ga) : "memory");
  }
}
static __device__ __forceinline__ void wait_async0() {
  asm volatile("s_wait_asynccnt 0x0" ::: "memory");
}

// ---------------- elementwise / scatter kernels ----------------

__global__ void zero_f32(float* __restrict__ p, int n) {
  int i = blockIdx.x * blockDim.x + threadIdx.x;
  if (i < n) p[i] = 0.0f;
}

__global__ void transpose_to_f16(const float* __restrict__ src,
                                 _Float16* __restrict__ dst, int K, int NC) {
  int t = blockIdx.x * blockDim.x + threadIdx.x;
  if (t >= K * NC) return;
  int n = t / K, k = t - n * K;        // dst[n*K + k] = src[k*NC + n]
  dst[t] = (_Float16)src[k * NC + n];
}

__global__ void count_nodes(const int* __restrict__ batch, float* __restrict__ cnt) {
  int i = blockIdx.x * blockDim.x + threadIdx.x;
  if (i < NN) atomicAdd(cnt + batch[i], 1.0f);
}

__global__ void init_aggr(const float* __restrict__ h, const float* __restrict__ eps,
                          int l, float* __restrict__ a) {
  int t = blockIdx.x * blockDim.x + threadIdx.x;
  if (t >= NN * HID) return;
  a[t] = (1.0f + eps[l]) * h[t];
}

// 4 lanes per edge; each lane moves 32 features (8x float4 gather + 32 atomics).
__global__ void edge_scatter(const int* __restrict__ ei, const float* __restrict__ h,
                             float* __restrict__ aggr) {
  unsigned t = blockIdx.x * blockDim.x + threadIdx.x;
  if (t >= (unsigned)NE * 4u) return;
  int e = t >> 2, q = t & 3;
  int s = ei[e];
  int d = ei[NE + e];
  const float4* hp = (const float4*)(h + (long)s * HID + q * 32);
  float* ap = aggr + (long)d * HID + q * 32;
#pragma unroll
  for (int i = 0; i < 8; ++i) {
    float4 v = hp[i];
    atomicAdd(ap + 4 * i + 0, v.x);
    atomicAdd(ap + 4 * i + 1, v.y);
    atomicAdd(ap + 4 * i + 2, v.z);
    atomicAdd(ap + 4 * i + 3, v.w);
  }
}

// batch is sorted: accumulate runs privately, flush one atomic per run.
// Block covers 64 rows; thread owns column (tid&127), row parity (tid>>7).
__global__ void seg_sum_sorted(const float* __restrict__ h,
                               const int* __restrict__ batch,
                               float* __restrict__ msum) {
  const int c = threadIdx.x & 127;
  const int rp = threadIdx.x >> 7;
  const long r0 = (long)blockIdx.x * 64;
  float acc = 0.0f; int curg = -1;
  for (int i = 0; i < 32; ++i) {
    long r = r0 + i * 2 + rp;
    if (r >= NN) break;
    int g = batch[r];
    if (g != curg) {
      if (curg >= 0) atomicAdd(msum + curg * HID + c, acc);
      curg = g; acc = 0.0f;
    }
    acc += h[r * HID + c];
  }
  if (curg >= 0) atomicAdd(msum + curg * HID + c, acc);
}

__global__ void sub_var_sorted(const float* __restrict__ h,
                               const int* __restrict__ batch,
                               const float* __restrict__ msum,
                               const float* __restrict__ cnt,
                               const float* __restrict__ scale,
                               float* __restrict__ sub,
                               float* __restrict__ vsum) {
  const int c = threadIdx.x & 127;
  const int rp = threadIdx.x >> 7;
  const float sc = scale[c];
  const long r0 = (long)blockIdx.x * 64;
  float acc = 0.0f, ms = 0.0f; int curg = -1;
  for (int i = 0; i < 32; ++i) {
    long r = r0 + i * 2 + rp;
    if (r >= NN) break;
    int g = batch[r];
    if (g != curg) {
      if (curg >= 0) atomicAdd(vsum + curg * HID + c, acc);
      curg = g; acc = 0.0f;
      ms = msum[g * HID + c] / fmaxf(cnt[g], 1.0f) * sc;
    }
    float s = h[r * HID + c] - ms;
    sub[r * HID + c] = s;
    acc += s * s;
  }
  if (curg >= 0) atomicAdd(vsum + curg * HID + c, acc);
}

__global__ void gnorm_relu(const float* __restrict__ sub, const int* __restrict__ batch,
                           const float* __restrict__ vsum, const float* __restrict__ cnt,
                           const float* __restrict__ w, const float* __restrict__ b,
                           float* __restrict__ h) {
  int t = blockIdx.x * blockDim.x + threadIdx.x;
  if (t >= NN * HID) return;
  int row = t >> 7, col = t & 127;
  int g = batch[row];
  float c = fmaxf(cnt[g], 1.0f);
  float sd = sqrtf(vsum[g * HID + col] / c + 1e-8f);
  float v = w[col] * sub[t] / sd + b[col];
  h[t] = fmaxf(v, 0.0f);
}

// ---------------- fused WMMA MLP kernels ----------------
// 256 threads = 8 waves; block handles 128 rows; wave owns 16 rows x 128 cols.
// Two N-tiles per iteration: 8 B-frags preloaded -> one dscnt wait, two
// independent WMMA accumulation chains.

__global__ void __launch_bounds__(256)
gin_conv_mlp(const float* __restrict__ X, const _Float16* __restrict__ W1t,
             const float* __restrict__ B1, const _Float16* __restrict__ W2t,
             const float* __restrict__ B2, float* __restrict__ Y) {
  extern __shared__ char smem[];
  _Float16* w1  = (_Float16*)smem;        // 128x128 (n-major)   LDS @ 0
  _Float16* w2  = w1 + 128 * 128;         // 128x128             LDS @ 32768
  _Float16* mid = w2 + 128 * 128;         // 128x128 f16         LDS @ 65536
  float* b1s = (float*)(mid + 128 * 128); // 128
  float* b2s = b1s + 128;                 // 128

  const int tid = threadIdx.x;
  async_copy_16(0u,     (const char*)W1t, 32768, tid);
  async_copy_16(32768u, (const char*)W2t, 32768, tid);
  if (tid < 128) { b1s[tid] = B1[tid]; b2s[tid] = B2[tid]; }
  wait_async0();
  __syncthreads();

  const int lane = tid & 31;
  const int wave = tid >> 5;
  const int lm = lane & 15;
  const int hi = lane >> 4;
  const long rowbase = (long)blockIdx.x * 128 + wave * 16;

  long arow = rowbase + lm;
  if (arow >= NN) arow = NN - 1;          // padded rows compute garbage, never stored
  const float* xr = X + arow * HID;

  v16h a[4];
#pragma unroll
  for (int k = 0; k < 4; ++k) a[k] = load_a_f32(xr, k * 32, hi);

  const int r0 = wave * 16 + hi * 8;
#pragma unroll
  for (int n = 0; n < 8; n += 2) {
    const int c0i = n * 16 + lm, c1i = c0i + 16;
    v16h b0[4], b1[4];
#pragma unroll
    for (int k = 0; k < 4; ++k) {
      b0[k] = *(const v16h*)(w1 + c0i * 128 + k * 32 + hi * 16);
      b1[k] = *(const v16h*)(w1 + c1i * 128 + k * 32 + hi * 16);
    }
    const float p0 = b1s[c0i], p1 = b1s[c1i];
    v8f c0 = {p0, p0, p0, p0, p0, p0, p0, p0};
    v8f c1 = {p1, p1, p1, p1, p1, p1, p1, p1};
#pragma unroll
    for (int k = 0; k < 4; ++k) {
      c0 = __builtin_amdgcn_wmma_f32_16x16x32_f16(false, a[k], false, b0[k],
                                                  (short)0, c0, false, false);
      c1 = __builtin_amdgcn_wmma_f32_16x16x32_f16(false, a[k], false, b1[k],
                                                  (short)0, c1, false, false);
    }
#pragma unroll
    for (int v = 0; v < 8; ++v) {
      mid[(r0 + v) * 128 + c0i] = (_Float16)fmaxf(c0[v], 0.0f);
      mid[(r0 + v) * 128 + c1i] = (_Float16)fmaxf(c1[v], 0.0f);
    }
  }
  __syncthreads();

  const _Float16* mrow = mid + (wave * 16 + lm) * 128;
  v16h a2[4];
#pragma unroll
  for (int k = 0; k < 4; ++k) a2[k] = load_a_lds(mrow, k * 32, hi);

  const long rg = rowbase + hi * 8;
#pragma unroll
  for (int n = 0; n < 8; n += 2) {
    const int c0i = n * 16 + lm, c1i = c0i + 16;
    v16h b0[4], b1[4];
#pragma unroll
    for (int k = 0; k < 4; ++k) {
      b0[k] = *(const v16h*)(w2 + c0i * 128 + k * 32 + hi * 16);
      b1[k] = *(const v16h*)(w2 + c1i * 128 + k * 32 + hi * 16);
    }
    const float p0 = b2s[c0i], p1 = b2s[c1i];
    v8f c0 = {p0, p0, p0, p0, p0, p0, p0, p0};
    v8f c1 = {p1, p1, p1, p1, p1, p1, p1, p1};
#pragma unroll
    for (int k = 0; k < 4; ++k) {
      c0 = __builtin_amdgcn_wmma_f32_16x16x32_f16(false, a2[k], false, b0[k],
                                                  (short)0, c0, false, false);
      c1 = __builtin_amdgcn_wmma_f32_16x16x32_f16(false, a2[k], false, b1[k],
                                                  (short)0, c1, false, false);
    }
#pragma unroll
    for (int v = 0; v < 8; ++v) {
      const long rr = rg + v;
      if (rr < NN) {
        Y[rr * HID + c0i] = c0[v];
        Y[rr * HID + c1i] = c1[v];
      }
    }
  }
}

// proj MLP (128->128 relu ->64) + fused global-add-pool into d_out.
__global__ void __launch_bounds__(256)
gin_proj_pool(const float* __restrict__ X, const _Float16* __restrict__ W1t,
              const float* __restrict__ B1, const _Float16* __restrict__ W2t,
              const float* __restrict__ B2, const int* __restrict__ batch,
              float* __restrict__ out, int col_ofs) {
  extern __shared__ char smem[];
  _Float16* w1  = (_Float16*)smem;        // 128x128   LDS @ 0
  _Float16* w2  = w1 + 128 * 128;         // 64x128    LDS @ 32768
  _Float16* mid = w2 + 64 * 128;          // 128x128   LDS @ 49152
  float* b1s = (float*)(mid + 128 * 128); // 128
  float* b2s = b1s + 128;                 // 64

  const int tid = threadIdx.x;
  async_copy_16(0u,     (const char*)W1t, 32768, tid);
  async_copy_16(32768u, (const char*)W2t, 16384, tid);
  if (tid < 128) b1s[tid] = B1[tid];
  if (tid < 64)  b2s[tid] = B2[tid];
  wait_async0();
  __syncthreads();

  const int lane = tid & 31;
  const int wave = tid >> 5;
  const int lm = lane & 15;
  const int hi = lane >> 4;
  const long rowbase = (long)blockIdx.x * 128 + wave * 16;

  long arow = rowbase + lm;
  if (arow >= NN) arow = NN - 1;
  const float* xr = X + arow * HID;

  v16h a[4];
#pragma unroll
  for (int k = 0; k < 4; ++k) a[k] = load_a_f32(xr, k * 32, hi);

  const int r0 = wave * 16 + hi * 8;
#pragma unroll
  for (int n = 0; n < 8; n += 2) {
    const int c0i = n * 16 + lm, c1i = c0i + 16;
    v16h b0[4], b1[4];
#pragma unroll
    for (int k = 0; k < 4; ++k) {
      b0[k] = *(const v16h*)(w1 + c0i * 128 + k * 32 + hi * 16);
      b1[k] = *(const v16h*)(w1 + c1i * 128 + k * 32 + hi * 16);
    }
    const float p0 = b1s[c0i], p1 = b1s[c1i];
    v8f c0 = {p0, p0, p0, p0, p0, p0, p0, p0};
    v8f c1 = {p1, p1, p1, p1, p1, p1, p1, p1};
#pragma unroll
    for (int k = 0; k < 4; ++k) {
      c0 = __builtin_amdgcn_wmma_f32_16x16x32_f16(false, a[k], false, b0[k],
                                                  (short)0, c0, false, false);
      c1 = __builtin_amdgcn_wmma_f32_16x16x32_f16(false, a[k], false, b1[k],
                                                  (short)0, c1, false, false);
    }
#pragma unroll
    for (int v = 0; v < 8; ++v) {
      mid[(r0 + v) * 128 + c0i] = (_Float16)fmaxf(c0[v], 0.0f);
      mid[(r0 + v) * 128 + c1i] = (_Float16)fmaxf(c1[v], 0.0f);
    }
  }
  __syncthreads();

  const _Float16* mrow = mid + (wave * 16 + lm) * 128;
  v16h a2[4];
#pragma unroll
  for (int k = 0; k < 4; ++k) a2[k] = load_a_lds(mrow, k * 32, hi);

  const long rg = rowbase + hi * 8;
#pragma unroll
  for (int n = 0; n < 4; n += 2) {        // TGT = 64 -> 4 col tiles
    const int c0i = n * 16 + lm, c1i = c0i + 16;
    v16h b0[4], b1[4];
#pragma unroll
    for (int k = 0; k < 4; ++k) {
      b0[k] = *(const v16h*)(w2 + c0i * 128 + k * 32 + hi * 16);
      b1[k] = *(const v16h*)(w2 + c1i * 128 + k * 32 + hi * 16);
    }
    const float p0 = b2s[c0i], p1 = b2s[c1i];
    v8f c0 = {p0, p0, p0, p0, p0, p0, p0, p0};
    v8f c1 = {p1, p1, p1, p1, p1, p1, p1, p1};
#pragma unroll
    for (int k = 0; k < 4; ++k) {
      c0 = __builtin_amdgcn_wmma_f32_16x16x32_f16(false, a2[k], false, b0[k],
                                                  (short)0, c0, false, false);
      c1 = __builtin_amdgcn_wmma_f32_16x16x32_f16(false, a2[k], false, b1[k],
                                                  (short)0, c1, false, false);
    }
#pragma unroll
    for (int v = 0; v < 8; ++v) {
      const long rr = rg + v;
      if (rr < NN) {
        int g = batch[rr];
        float* o = out + (long)g * (NL * TGT) + col_ofs;
        atomicAdd(o + c0i, c0[v]);
        atomicAdd(o + c1i, c1[v]);
      }
    }
  }
}

// ---------------- host launch ----------------

extern "C" void kernel_launch(void* const* d_in, const int* in_sizes, int n_in,
                              void* d_out, int out_size, void* d_ws, size_t ws_size,
                              hipStream_t stream) {
  const float* x    = (const float*)d_in[0];
  const int*   ei   = (const int*)  d_in[1];
  const int*   batch= (const int*)  d_in[2];
  const float* cw1  = (const float*)d_in[3];
  const float* cb1  = (const float*)d_in[4];
  const float* cw2  = (const float*)d_in[5];
  const float* cb2  = (const float*)d_in[6];
  const float* eps  = (const float*)d_in[7];
  const float* gsc  = (const float*)d_in[8];
  const float* gwt  = (const float*)d_in[9];
  const float* gbi  = (const float*)d_in[10];
  const float* pw1  = (const float*)d_in[11];
  const float* pb1  = (const float*)d_in[12];
  const float* pw2  = (const float*)d_in[13];
  const float* pb2  = (const float*)d_in[14];
  float* out = (float*)d_out;

  // workspace partition (floats)
  float* hbuf = (float*)d_ws;                      // N*HID current features
  float* abuf = hbuf + (size_t)NN * HID;           // N*HID aggr / sub scratch
  float* msum = abuf + (size_t)NN * HID;           // G*HID
  float* vsum = msum + (size_t)NG * HID;           // G*HID
  float* cnts = vsum + (size_t)NG * HID;           // G
  _Float16* wt = (_Float16*)(cnts + NG);           // f16 transposed weights
  const size_t WSTRIDE = 57344;                    // halves per layer

  const int EW = (NN * HID + 255) / 256;           // 12.8M elementwise grid
  const int EDGE_BLK = (NE * 4 + 255) / 256;
  const int NODE_BLK = (NN + 255) / 256;
  const int SEG_BLK = (NN + 63) / 64;              // 64 rows per block
  const int MLP_BLK = (NN + 127) / 128;            // 782
  const size_t SM_CONV = (size_t)(128*128*2)*2 + 128*128*2 + 2*128*4; // 99328 B
  const size_t SM_PROJ = (size_t)128*128*2 + 64*128*2 + 128*128*2 + 128*4 + 64*4;

  // init accumulators / output
  zero_f32<<<(out_size + 255) / 256, 256, 0, stream>>>(out, out_size);
  zero_f32<<<(NG + 255) / 256, 256, 0, stream>>>(cnts, NG);
  count_nodes<<<NODE_BLK, 256, 0, stream>>>(batch, cnts);

  // pre-transpose + convert all weights to f16 [n][k]
  for (int l = 0; l < NL; ++l) {
    _Float16* wl = wt + (size_t)l * WSTRIDE;
    transpose_to_f16<<<64, 256, 0, stream>>>(cw1 + (size_t)l * FEAT * HID, wl,         FEAT, HID);
    transpose_to_f16<<<64, 256, 0, stream>>>(cw2 + (size_t)l * HID * HID,  wl + 16384, HID,  HID);
    transpose_to_f16<<<64, 256, 0, stream>>>(pw1 + (size_t)l * HID * HID,  wl + 32768, HID,  HID);
    transpose_to_f16<<<32, 256, 0, stream>>>(pw2 + (size_t)l * HID * TGT,  wl + 49152, HID,  TGT);
  }

  for (int l = 0; l < NL; ++l) {
    const float* hin = (l == 0) ? x : hbuf;
    _Float16* wl = wt + (size_t)l * WSTRIDE;

    // aggr = (1+eps)*h + segment_sum(h[src] -> dst)
    init_aggr<<<EW, 256, 0, stream>>>(hin, eps, l, abuf);
    edge_scatter<<<EDGE_BLK, 256, 0, stream>>>(ei, hin, abuf);

    // h = relu(aggr@W1+b1)@W2+b2   (WMMA, f16 in / f32 acc)
    gin_conv_mlp<<<MLP_BLK, 256, SM_CONV, stream>>>(
        abuf, wl, cb1 + l * HID, wl + 16384, cb2 + l * HID, hbuf);

    // GraphNorm + relu (sorted-run segment reductions)
    zero_f32<<<(2 * NG * HID + 255) / 256, 256, 0, stream>>>(msum, 2 * NG * HID);
    seg_sum_sorted<<<SEG_BLK, 256, 0, stream>>>(hbuf, batch, msum);
    sub_var_sorted<<<SEG_BLK, 256, 0, stream>>>(hbuf, batch, msum, cnts,
                                                gsc + l * HID, abuf, vsum);
    gnorm_relu<<<EW, 256, 0, stream>>>(abuf, batch, vsum, cnts,
                                       gwt + l * HID, gbi + l * HID, hbuf);

    // z = relu(h@P1+b1)@P2+b2, pooled per graph into out[:, l*TGT : (l+1)*TGT]
    gin_proj_pool<<<MLP_BLK, 256, SM_PROJ, stream>>>(
        hbuf, wl + 32768, pb1 + l * HID, wl + 49152, pb2 + l * TGT,
        batch, out, l * TGT);
  }
}
